// ReconstructorHGNN_25933012533353
// MI455X (gfx1250) — matrix operand order repaired and checked by reading, not compiled, not run on hardware
//
#include <hip/hip_runtime.h>

typedef __attribute__((ext_vector_type(16))) _Float16 v16h;
typedef __attribute__((ext_vector_type(4)))  _Float16 v4h;
typedef __attribute__((ext_vector_type(8)))  float    v8f;
typedef __attribute__((ext_vector_type(4)))  int      v4i;

// ---------------------------------------------------------------------------
// Async global->LDS copy (16B per lane). gfx1250 async path when available,
// plain vector LDS copy otherwise.
// Probe result: __builtin_amdgcn_global_load_async_to_lds_b128 exists with
// signature (addrspace(1) int4*, addrspace(3) int4*, imm offset, imm cpol).
// ---------------------------------------------------------------------------
#if __has_builtin(__builtin_amdgcn_global_load_async_to_lds_b128)
#define GCN_ASYNC_LDS 1
#endif

template <typename T>
__device__ __forceinline__ void copy16_g2lds(T* lds_dst, const T* g_src) {
#ifdef GCN_ASYNC_LDS
  auto gsrc = reinterpret_cast<__attribute__((address_space(1))) v4i*>(
      reinterpret_cast<uintptr_t>(g_src));
  auto ldst = reinterpret_cast<__attribute__((address_space(3))) v4i*>(
      reinterpret_cast<uintptr_t>(lds_dst));
  __builtin_amdgcn_global_load_async_to_lds_b128(gsrc, ldst, 0, 0);
#else
  __builtin_memcpy(lds_dst, g_src, 16);
#endif
}

__device__ __forceinline__ void wait_async_lds() {
#ifdef GCN_ASYNC_LDS
  asm volatile("s_wait_asynccnt 0x0" ::: "memory");
#endif
}

// ---------------------------------------------------------------------------
// Node init: deg = 1.0 (self loop), counts = 0
// ---------------------------------------------------------------------------
__global__ __launch_bounds__(256) void gcn_init_nodes(float* __restrict__ deg,
                                                      unsigned* __restrict__ counts, int n) {
  int i = blockIdx.x * 256 + threadIdx.x;
  if (i < n) { deg[i] = 1.0f; counts[i] = 0u; }
}

// ---------------------------------------------------------------------------
// Per-edge: deg[dst] += ew ; counts[dst] += 1   (edge_index: [0,E)=src, [E,2E)=dst)
// ---------------------------------------------------------------------------
__global__ __launch_bounds__(256) void gcn_edge_count(const int* __restrict__ ei,
                                                      const float* __restrict__ ew,
                                                      float* __restrict__ deg,
                                                      unsigned* __restrict__ counts, int E) {
  int e = blockIdx.x * 256 + threadIdx.x;
  if (e >= E) return;
  int d = ei[E + e];
  atomicAdd(&deg[d], ew[e]);
  atomicAdd(&counts[d], 1u);
}

// ---------------------------------------------------------------------------
// Single-block exclusive scan of counts -> rowptr
// ---------------------------------------------------------------------------
__global__ __launch_bounds__(1024) void gcn_scan(const unsigned* __restrict__ counts,
                                                 unsigned* __restrict__ rowptr, int n) {
  __shared__ unsigned sh[1024];
  __shared__ unsigned carry;
  int tid = threadIdx.x;
  if (tid == 0) carry = 0u;
  __syncthreads();
  for (int base = 0; base < n; base += 1024) {
    int i = base + tid;
    unsigned v = (i < n) ? counts[i] : 0u;
    sh[tid] = v;
    __syncthreads();
    for (int off = 1; off < 1024; off <<= 1) {
      unsigned t = (tid >= off) ? sh[tid - off] : 0u;
      __syncthreads();
      sh[tid] += t;
      __syncthreads();
    }
    unsigned c = carry;
    if (i < n) rowptr[i] = c + sh[tid] - v;   // exclusive
    __syncthreads();
    if (tid == 1023) carry = c + sh[1023];
    __syncthreads();
  }
  if (tid == 0) rowptr[n] = carry;
}

// ---------------------------------------------------------------------------
// dis = rsqrt(deg) ; cursor = rowptr
// ---------------------------------------------------------------------------
__global__ __launch_bounds__(256) void gcn_finalize_nodes(const float* __restrict__ deg,
                                                          const unsigned* __restrict__ rowptr,
                                                          float* __restrict__ dis,
                                                          unsigned* __restrict__ cursor, int n) {
  int i = blockIdx.x * 256 + threadIdx.x;
  if (i < n) {
    dis[i] = rsqrtf(deg[i]);
    cursor[i] = rowptr[i];
  }
}

// ---------------------------------------------------------------------------
// CSR fill: col[p] = src ; val[p] = dis[src]*ew*dis[dst]
// ---------------------------------------------------------------------------
__global__ __launch_bounds__(256) void gcn_csr_fill(const int* __restrict__ ei,
                                                    const float* __restrict__ ew,
                                                    const float* __restrict__ dis,
                                                    unsigned* __restrict__ cursor,
                                                    unsigned* __restrict__ col,
                                                    float* __restrict__ val, int E) {
  int e = blockIdx.x * 256 + threadIdx.x;
  if (e >= E) return;
  int s = ei[e];
  int d = ei[E + e];
  unsigned p = atomicAdd(&cursor[d], 1u);
  col[p] = (unsigned)s;
  val[p] = dis[s] * ew[e] * dis[d];
}

// ---------------------------------------------------------------------------
// f32 -> f16 (vectorized x4)
// ---------------------------------------------------------------------------
__global__ __launch_bounds__(256) void gcn_cvt_f16(const float* __restrict__ src,
                                                   _Float16* __restrict__ dst, int n4) {
  int i = blockIdx.x * 256 + threadIdx.x;
  if (i < n4) {
    float4 v = ((const float4*)src)[i];
    v4h o = {(_Float16)v.x, (_Float16)v.y, (_Float16)v.z, (_Float16)v.w};
    ((v4h*)dst)[i] = o;
  }
}

// ---------------------------------------------------------------------------
// W[128][128] f32 -> Wt[n][k] f16 (transposed)
// ---------------------------------------------------------------------------
__global__ __launch_bounds__(256) void gcn_wt_f16(const float* __restrict__ W,
                                                  _Float16* __restrict__ Wt) {
  int idx = blockIdx.x * 256 + threadIdx.x;
  if (idx < 128 * 128) {
    int nn = idx >> 7, kk = idx & 127;
    Wt[idx] = (_Float16)W[kk * 128 + nn];
  }
}

// ---------------------------------------------------------------------------
// Y[M][128] = A[M][128] @ W, A in f16 row-major, W pre-transposed f16 [n][k].
// 256 threads = 8 waves. Block computes a 64x128 band: wave w owns columns
// [16w,16w+16), 4 row-tiles -> 4 accumulators, 16 v_wmma per wave.
// LDS rows padded to 136 halves: every b128 fragment read is conflict-free.
// ---------------------------------------------------------------------------
#define LPAD 136

__global__ __launch_bounds__(256) void gcn_gemm_wmma(const _Float16* __restrict__ A,
                                                     const _Float16* __restrict__ Wt,
                                                     float* __restrict__ Y, int M) {
  __shared__ __align__(16) _Float16 shW[128][LPAD];
  __shared__ __align__(16) _Float16 shA[64][LPAD];
  const int tid  = threadIdx.x;
  const int wave = tid >> 5;
  const int lane = tid & 31;
  const int m    = lane & 15;
  const int g    = lane >> 4;
  const int row0 = blockIdx.x * 64;

  // stage Wt: 128 rows x 128 halves = 2048 x 16B chunks (8/thread)
#pragma unroll
  for (int it = 0; it < 8; ++it) {
    int c = it * 256 + tid;
    int r = c >> 4, q = c & 15;
    copy16_g2lds(&shW[r][q * 8], Wt + r * 128 + q * 8);
  }
  // stage A band: 64 rows x 128 halves = 1024 x 16B chunks (4/thread)
#pragma unroll
  for (int it = 0; it < 4; ++it) {
    int c = it * 256 + tid;
    int r = c >> 4, q = c & 15;
    int rr = row0 + r; if (rr >= M) rr = M - 1;
    copy16_g2lds(&shA[r][q * 8], A + (size_t)rr * 128 + q * 8);
  }
  wait_async_lds();
  __syncthreads();

  v8f acc[4] = {};
#pragma unroll
  for (int ks = 0; ks < 4; ++ks) {
    // B fragment (32x16): lane group g holds K = 8g+e / 16+8g+(e-8): 2x16B chunks.
    v16h b;
    const _Float16* pb = &shW[wave * 16 + m][ks * 32 + 8 * g];
#pragma unroll
    for (int e2 = 0; e2 < 8; ++e2) { b[e2] = pb[e2]; b[8 + e2] = pb[16 + e2]; }
#pragma unroll
    for (int t = 0; t < 4; ++t) {
      // A fragment (16x32): group g holds K = 8g+e / 16+8g+(e-8): 2x16B chunks.
      v16h a;
      const _Float16* pa0 = &shA[t * 16 + m][ks * 32 + 8 * g];
      const _Float16* pa1 = pa0 + 16;
#pragma unroll
      for (int e2 = 0; e2 < 8; ++e2) { a[e2] = pa0[e2]; a[8 + e2] = pa1[e2]; }
      acc[t] = __builtin_amdgcn_wmma_f32_16x16x32_f16(
          false, a, false, b, (short)0, acc[t], false, false);
    }
  }

  // C/D layout: VGPR r -> row (r + 8g), col = 16*wave + (lane&15)
  const int n = wave * 16 + m;
#pragma unroll
  for (int t = 0; t < 4; ++t) {
#pragma unroll
    for (int r = 0; r < 8; ++r) {
      int row = row0 + t * 16 + r + 8 * g;
      if (row < M) Y[(size_t)row * 128 + n] = acc[t][r];
    }
  }
}

// ---------------------------------------------------------------------------
// Aggregation: one wave per node, lane owns a float4 chunk (32*4 = 128 dims).
// acc = H[node]*dis^2 + sum_e val[e]*H[col[e]] + bias ; optional relu.
// Output: f32 (Yf) or f16 (Yh) depending on which pointer is non-null.
// ---------------------------------------------------------------------------
__global__ __launch_bounds__(256) void gcn_aggregate(const float* __restrict__ H,
                                                     const unsigned* __restrict__ rowptr,
                                                     const unsigned* __restrict__ col,
                                                     const float* __restrict__ val,
                                                     const float* __restrict__ dis,
                                                     const float* __restrict__ bias,
                                                     float* __restrict__ Yf,
                                                     _Float16* __restrict__ Yh,
                                                     int n, int relu) {
  int wave = threadIdx.x >> 5;
  int lane = threadIdx.x & 31;
  int node = blockIdx.x * 8 + wave;
  if (node >= n) return;

  const float4* Hv = (const float4*)H;
  float d = dis[node];
  float self = d * d;
  float4 h0 = Hv[node * 32 + lane];
  float ax = h0.x * self, ay = h0.y * self, az = h0.z * self, aw = h0.w * self;

  unsigned beg = rowptr[node], end = rowptr[node + 1];
  for (unsigned e = beg; e < end; ++e) {
    if (e + 4 < end)
      __builtin_prefetch(&Hv[(size_t)col[e + 4] * 32 + lane], 0, 0);  // global_prefetch_b8
    unsigned s = col[e];
    float w = val[e];
    float4 hv = Hv[(size_t)s * 32 + lane];
    ax = fmaf(w, hv.x, ax);
    ay = fmaf(w, hv.y, ay);
    az = fmaf(w, hv.z, az);
    aw = fmaf(w, hv.w, aw);
  }

  float4 bv = ((const float4*)bias)[lane];
  ax += bv.x; ay += bv.y; az += bv.z; aw += bv.w;
  if (relu) {
    ax = fmaxf(ax, 0.0f); ay = fmaxf(ay, 0.0f);
    az = fmaxf(az, 0.0f); aw = fmaxf(aw, 0.0f);
  }
  if (Yh) {
    v4h o = {(_Float16)ax, (_Float16)ay, (_Float16)az, (_Float16)aw};
    ((v4h*)Yh)[node * 32 + lane] = o;
  } else {
    float4 o; o.x = ax; o.y = ay; o.z = az; o.w = aw;
    ((float4*)Yf)[node * 32 + lane] = o;
  }
}

// ---------------------------------------------------------------------------
// Launch
// ---------------------------------------------------------------------------
extern "C" void kernel_launch(void* const* d_in, const int* in_sizes, int n_in,
                              void* d_out, int out_size, void* d_ws, size_t ws_size,
                              hipStream_t stream) {
  const float* x  = (const float*)d_in[0];
  const int*   ei = (const int*)d_in[1];
  const float* ew = (const float*)d_in[2];
  const float* W1 = (const float*)d_in[3];
  const float* b1 = (const float*)d_in[4];
  const float* W2 = (const float*)d_in[5];
  const float* b2 = (const float*)d_in[6];
  float* out = (float*)d_out;

  const int N = in_sizes[0] / 128;   // 50000
  const int E = in_sizes[2];         // 800000
  (void)n_in; (void)out_size; (void)ws_size;

  char* ws = (char*)d_ws;
  size_t off = 0;
  auto take = [&](size_t bytes) -> char* {
    char* p = ws + off;
    off = (off + bytes + 255) & ~(size_t)255;
    return p;
  };
  float*     deg    = (float*)take((size_t)N * 4);
  float*     dis    = (float*)take((size_t)N * 4);
  unsigned*  counts = (unsigned*)take((size_t)N * 4);
  unsigned*  rowptr = (unsigned*)take((size_t)(N + 1) * 4);
  unsigned*  cursor = (unsigned*)take((size_t)N * 4);
  unsigned*  col    = (unsigned*)take((size_t)E * 4);
  float*     val    = (float*)take((size_t)E * 4);
  float*     t1     = (float*)take((size_t)N * 128 * 4);   // GEMM out (both layers)
  _Float16*  xh     = (_Float16*)take((size_t)N * 128 * 2);
  _Float16*  t2h    = (_Float16*)take((size_t)N * 128 * 2); // layer-1 activation (f16)
  _Float16*  wt1    = (_Float16*)take((size_t)128 * 128 * 2);
  _Float16*  wt2    = (_Float16*)take((size_t)128 * 128 * 2);

  const int nb = (N + 255) / 256;
  const int eb = (E + 255) / 256;
  const int gb = (N + 63) / 64;            // 64-row GEMM bands
  const int ab = (N + 7) / 8;              // 8 nodes (waves) per block
  const int cb = ((N * 128 / 4) + 255) / 256;

  // Graph preprocessing (CSR by dst + symmetric norm)
  gcn_init_nodes<<<nb, 256, 0, stream>>>(deg, counts, N);
  gcn_edge_count<<<eb, 256, 0, stream>>>(ei, ew, deg, counts, E);
  gcn_scan<<<1, 1024, 0, stream>>>(counts, rowptr, N);
  gcn_finalize_nodes<<<nb, 256, 0, stream>>>(deg, rowptr, dis, cursor, N);
  gcn_csr_fill<<<eb, 256, 0, stream>>>(ei, ew, dis, cursor, col, val, E);

  // f16 operand prep
  gcn_cvt_f16<<<cb, 256, 0, stream>>>(x, xh, N * 128 / 4);
  gcn_wt_f16<<<64, 256, 0, stream>>>(W1, wt1);
  gcn_wt_f16<<<64, 256, 0, stream>>>(W2, wt2);

  // Layer 1: t1 = x@W1 ; t2h = f16(relu(agg(t1) + b1))
  gcn_gemm_wmma<<<gb, 256, 0, stream>>>(xh, wt1, t1, N);
  gcn_aggregate<<<ab, 256, 0, stream>>>(t1, rowptr, col, val, dis, b1,
                                        nullptr, t2h, N, 1);

  // Layer 2: t1 = t2h@W2 ; out = agg(t1) + b2
  gcn_gemm_wmma<<<gb, 256, 0, stream>>>(t2h, wt2, t1, N);
  gcn_aggregate<<<ab, 256, 0, stream>>>(t1, rowptr, col, val, dis, b2,
                                        out, nullptr, N, 0);
}
